// FMoE_49804440764686
// MI455X (gfx1250) — matrix operand order repaired and checked by reading, not compiled, not run on hardware
//
#include <hip/hip_runtime.h>
#include <stdint.h>

#define N_TOK  4096
#define DMODEL 512
#define HDIM   2048
#define NEXP   8

typedef __attribute__((ext_vector_type(4)))  float    v4f;
typedef __attribute__((ext_vector_type(4)))  int      v4i;
typedef __attribute__((ext_vector_type(8)))  int      v8i;
typedef __attribute__((ext_vector_type(8)))  float    v8f;
typedef __attribute__((ext_vector_type(4)))  unsigned su4;
typedef __attribute__((ext_vector_type(8)))  unsigned su8;
typedef __attribute__((ext_vector_type(16))) __bf16   v16bf;

__device__ inline unsigned short f2bf(float f) {
  union { float f; unsigned u; } v; v.f = f;
  unsigned r = v.u + 0x7FFFu + ((v.u >> 16) & 1u);  // round-to-nearest-even
  return (unsigned short)(r >> 16);
}
__device__ inline unsigned pack2bf(float a, float b) {
  return (unsigned)f2bf(a) | ((unsigned)f2bf(b) << 16);
}
__device__ inline float gelu_tanh(float x) {
  const float c0 = 0.7978845608028654f, c1 = 0.044715f;
  float x3 = x * x * x;
  return 0.5f * x * (1.0f + tanhf(c0 * (x + c1 * x3)));
}
__device__ inline v16bf make_frag(v4i lo, v4i hi) {
  v8i t;
  t[0] = lo[0]; t[1] = lo[1]; t[2] = lo[2]; t[3] = lo[3];
  t[4] = hi[0]; t[5] = hi[1]; t[6] = hi[2]; t[7] = hi[3];
  return __builtin_bit_cast(v16bf, t);
}
__device__ inline v8f v8f_zero() {
  v8f z;
#pragma unroll
  for (int i = 0; i < 8; ++i) z[i] = 0.0f;
  return z;
}

// ---- CDNA5 async global->LDS copy (ASYNCcnt path) -------------------------
__device__ inline void async_copy_b128(unsigned ldsoff, const void* gaddr) {
  asm volatile("global_load_async_to_lds_b128 %0, %1, off"
               :: "v"(ldsoff), "v"(gaddr) : "memory");
}
__device__ inline void wait_async0() {
  asm volatile("s_wait_asynccnt 0x0" ::: "memory");
}
// ---- CDNA5 Tensor Data Mover (TENSORcnt path) -----------------------------
// Loads a [128 rows x 32 cols] bf16 tile (rows strided by row_len elements)
// into LDS at byte offset `lds`. 2-D descriptor: groups 0+1 only.
__device__ inline void tdm_load_tile_128x32_bf16(unsigned lds,
                                                 const unsigned short* gaddr,
                                                 unsigned row_len,   // tensor_dim0 & dim0 stride (elems)
                                                 unsigned nrows) {   // tensor_dim1
  uint64_t ga = (uint64_t)(uintptr_t)gaddr;
  su4 g0;
  g0[0] = 1u;                                            // count=1, user descriptor
  g0[1] = lds;                                           // lds_addr (bytes)
  g0[2] = (unsigned)ga;                                  // global_addr[31:0]
  g0[3] = ((unsigned)(ga >> 32) & 0x01FFFFFFu) | (2u << 30);  // addr[56:32] | type=2
  su8 g1;
  g1[0] = 1u << 16;                                      // data_size=1 (2 bytes)
  g1[1] = (row_len & 0xFFFFu) << 16;                     // tensor_dim0[15:0] @bit48
  g1[2] = (row_len >> 16) | ((nrows & 0xFFFFu) << 16);   // dim0[31:16] | dim1[15:0]
  g1[3] = (nrows >> 16) | (32u << 16);                   // dim1[31:16] | tile_dim0=32
  g1[4] = 128u;                                          // tile_dim1=128, tile_dim2=0
  g1[5] = row_len;                                       // tensor_dim0_stride[31:0]
  g1[6] = 0u;                                            // stride[47:32] | dim1_stride lo
  g1[7] = 0u;
  asm volatile("tensor_load_to_lds %0, %1" :: "s"(g0), "s"(g1) : "memory");
}
__device__ inline void wait_tensor0() {
  asm volatile("s_wait_tensorcnt 0x0" ::: "memory");
}

// ---------------- one-time f32 -> bf16 convert (inp) -----------------------
__global__ __launch_bounds__(256)
void convert_x(const float* __restrict__ src, unsigned short* __restrict__ dst) {
  size_t idx = ((size_t)blockIdx.x * 256 + threadIdx.x) * 8;
  v4f a = *(const v4f*)(src + idx);
  v4f b = *(const v4f*)(src + idx + 4);
  v4i p;
  p[0] = (int)pack2bf(a[0], a[1]);
  p[1] = (int)pack2bf(a[2], a[3]);
  p[2] = (int)pack2bf(b[0], b[1]);
  p[3] = (int)pack2bf(b[2], b[3]);
  *(v4i*)(dst + idx) = p;
}

// ---------------- one-time transpose+convert: [R][C] f32 -> [C][R] bf16 ----
__global__ __launch_bounds__(256)
void transpose_cvt(const float* __restrict__ src, unsigned short* __restrict__ dst,
                   int R, int C) {
  __shared__ unsigned short T[32][33];   // +1 pad: no bank conflicts
  const size_t eoff = (size_t)blockIdx.z * R * C;
  src += eoff; dst += eoff;
  const int t = threadIdx.x;
  const int c0 = blockIdx.x * 32, r0 = blockIdx.y * 32;
  {
    int r = t >> 3, cg = (t & 7) << 2;
    v4f v = *(const v4f*)(src + (size_t)(r0 + r) * C + c0 + cg);
#pragma unroll
    for (int i = 0; i < 4; ++i) T[r][cg + i] = f2bf(v[i]);
  }
  __syncthreads();
  {
    int c = t >> 3, rg = (t & 7) << 2;
    unsigned* dp = (unsigned*)(dst + (size_t)(c0 + c) * R + r0 + rg);
    dp[0] = (unsigned)T[rg + 0][c] | ((unsigned)T[rg + 1][c] << 16);
    dp[1] = (unsigned)T[rg + 2][c] | ((unsigned)T[rg + 3][c] << 16);
  }
}

// ---------------- Gate: logits -> top-2 -> softmax -> combine[N][E] --------
__global__ __launch_bounds__(256)
void moe_gate(const float* __restrict__ inp, const float* __restrict__ gw,
              const float* __restrict__ gb, float* __restrict__ combine) {
  const int lane  = threadIdx.x & 31;
  const int token = blockIdx.x * 8 + (threadIdx.x >> 5);
  float acc[NEXP];
#pragma unroll
  for (int e = 0; e < NEXP; ++e) acc[e] = 0.0f;
  for (int d = lane; d < DMODEL; d += 32) {
    float x = inp[(size_t)token * DMODEL + d];
    v4f g0 = *(const v4f*)(gw + (size_t)d * NEXP);
    v4f g1 = *(const v4f*)(gw + (size_t)d * NEXP + 4);
    acc[0] += x * g0[0]; acc[1] += x * g0[1]; acc[2] += x * g0[2]; acc[3] += x * g0[3];
    acc[4] += x * g1[0]; acc[5] += x * g1[1]; acc[6] += x * g1[2]; acc[7] += x * g1[3];
  }
#pragma unroll
  for (int off = 16; off > 0; off >>= 1)
#pragma unroll
    for (int e = 0; e < NEXP; ++e) acc[e] += __shfl_xor(acc[e], off, 32);
  if (lane == 0) {
#pragma unroll
    for (int e = 0; e < NEXP; ++e) acc[e] += gb[e];
    int i1 = 0; float m1 = acc[0];
#pragma unroll
    for (int e = 1; e < NEXP; ++e) if (acc[e] > m1) { m1 = acc[e]; i1 = e; }
    int i2 = 0; float m2 = -3.4e38f;
#pragma unroll
    for (int e = 0; e < NEXP; ++e) if (e != i1 && acc[e] > m2) { m2 = acc[e]; i2 = e; }
    float e2 = __expf(m2 - m1);          // m2 <= m1, stable
    float s1 = 1.0f / (1.0f + e2);
    float s2 = 1.0f - s1;
#pragma unroll
    for (int e = 0; e < NEXP; ++e) {
      float w = (e == i1) ? s1 : ((e == i2) ? s2 : 0.0f);
      combine[(size_t)token * NEXP + e] = w;
    }
  }
}

// ======================= shared GEMM inner machinery =======================
// WG tile 128x128, 8 waves, wave tile 32x64 (2x4 WMMA). A and B both bf16:
//   A [M][K] row-major -> LDS Als[row][k] via async b128 copies
//   B pre-transposed [N][K] row-major -> LDS Bls[col][k] via one TDM load
struct Frags { v16bf a[2], b[4]; };

__device__ inline void load_tiles(const unsigned short* Aglob, int lda,
                                  const unsigned short* Bglob, int ldb, int nrowsB,
                                  unsigned short* Als, unsigned short* Bls,
                                  int t, int wid) {
  // A tile: 128x32 bf16, 512 async b128 transfers, 2 per thread.
#pragma unroll
  for (int j = 0; j < 2; ++j) {
    int seg = t + j * 256;           // 0..511
    int r = seg >> 2;                // 4 x 8-short segments per row
    int cg = (seg & 3) << 3;
    async_copy_b128((unsigned)(uintptr_t)&Als[r * 32 + cg],
                    (const void*)(Aglob + (size_t)r * lda + cg));
  }
  // B tile: one Tensor-DMA descriptor moves the whole 128x32 tile.
  if (wid == 0)
    tdm_load_tile_128x32_bf16((unsigned)(uintptr_t)&Bls[0], Bglob,
                              (unsigned)ldb, (unsigned)nrowsB);
}

__device__ inline void mma_tiles(const unsigned short* Als, const unsigned short* Bls,
                                 int lane, int waveM, int waveN, v8f acc[2][4]) {
  v16bf af[2], bfr[4];
#pragma unroll
  for (int m = 0; m < 2; ++m) {
    int r = waveM * 32 + m * 16 + (lane & 15);
    int kh = (lane >> 4) << 3;
    v4i lo = *(const v4i*)&Als[r * 32 + kh];
    v4i hi = *(const v4i*)&Als[r * 32 + kh + 16];
    af[m] = make_frag(lo, hi);
  }
#pragma unroll
  for (int n = 0; n < 4; ++n) {
    int c = waveN * 64 + n * 16 + (lane & 15);
    int kg = (lane >> 4) << 3;
    v4i lo = *(const v4i*)&Bls[c * 32 + kg];
    v4i hi = *(const v4i*)&Bls[c * 32 + kg + 16];
    bfr[n] = make_frag(lo, hi);
  }
#pragma unroll
  for (int m = 0; m < 2; ++m)
#pragma unroll
    for (int n = 0; n < 4; ++n)
      acc[m][n] = __builtin_amdgcn_wmma_f32_16x16x32_bf16(
          false, af[m], false, bfr[n], (short)0, acc[m][n], false, false);
}

// ---------------- FFN layer 1: h = gelu(X @ W1 + b1) -----------------------
__global__ __launch_bounds__(256)
void moe_ffn1(const unsigned short* __restrict__ X,    // x bf16 [4096][512]
              const unsigned short* __restrict__ W1t,  // w1t bf16 [2048][512]
              const float* __restrict__ bias,          // b1[e] [2048]
              unsigned short* __restrict__ Hout)       // h bf16 [4096][2048]
{
  __shared__ __align__(16) unsigned short Als[128 * 32];
  __shared__ __align__(16) unsigned short Bls[128 * 32];
  const int t = threadIdx.x;
  const int lane = t & 31;
  const int wid = __builtin_amdgcn_readfirstlane(t >> 5);  // scalar wave id
  const int waveM = wid & 3, waveN = wid >> 2;
  const int rowBase = blockIdx.y * 128;
  const int colBase = blockIdx.x * 128;

  v8f acc[2][4];
#pragma unroll
  for (int m = 0; m < 2; ++m)
#pragma unroll
    for (int n = 0; n < 4; ++n) acc[m][n] = v8f_zero();

  const int KT = DMODEL / 32;
  for (int kt = 0; kt < KT; ++kt) {
    const int kBase = kt * 32;
    load_tiles(X + (size_t)rowBase * DMODEL + kBase, DMODEL,
               W1t + (size_t)colBase * DMODEL + kBase, DMODEL, HDIM,
               Als, Bls, t, wid);
    if (kt + 1 < KT)
      __builtin_prefetch(X + (size_t)(rowBase + (t >> 1)) * DMODEL + kBase + 32, 0, 0);
    if (wid == 0) wait_tensor0();
    wait_async0();
    __syncthreads();
    mma_tiles(Als, Bls, lane, waveM, waveN, acc);
    __syncthreads();
  }

  // Epilogue: bias + GELU, store bf16 h.
#pragma unroll
  for (int m = 0; m < 2; ++m)
#pragma unroll
    for (int n = 0; n < 4; ++n) {
      int col = colBase + waveN * 64 + n * 16 + (lane & 15);
      float b = bias[col];
#pragma unroll
      for (int i = 0; i < 8; ++i) {
        int row = rowBase + waveM * 32 + m * 16 + i + ((lane >> 4) << 3);
        float x = acc[m][n][i] + b;
        Hout[(size_t)row * HDIM + col] = f2bf(gelu_tanh(x));
      }
    }
}

// ---------------- FFN layer 2: out (+)= combine[:,e] * (h @ W2 + b2) -------
__global__ __launch_bounds__(256)
void moe_ffn2(const unsigned short* __restrict__ Hin,  // h bf16 [4096][2048]
              const unsigned short* __restrict__ W2t,  // w2t bf16 [512][2048]
              const float* __restrict__ bias,          // b2[e] [512]
              const float* __restrict__ combine,       // [4096][8]
              int expert, int accum,
              float* __restrict__ Out)                 // [4096][512] f32
{
  __shared__ __align__(16) unsigned short Als[128 * 32];
  __shared__ __align__(16) unsigned short Bls[128 * 32];
  const int t = threadIdx.x;
  const int lane = t & 31;
  const int wid = __builtin_amdgcn_readfirstlane(t >> 5);
  const int waveM = wid & 3, waveN = wid >> 2;
  const int rowBase = blockIdx.y * 128;
  const int colBase = blockIdx.x * 128;

  v8f acc[2][4];
#pragma unroll
  for (int m = 0; m < 2; ++m)
#pragma unroll
    for (int n = 0; n < 4; ++n) acc[m][n] = v8f_zero();

  const int KT = HDIM / 32;
  for (int kt = 0; kt < KT; ++kt) {
    const int kBase = kt * 32;
    load_tiles(Hin + (size_t)rowBase * HDIM + kBase, HDIM,
               W2t + (size_t)colBase * HDIM + kBase, HDIM, DMODEL,
               Als, Bls, t, wid);
    if (kt + 1 < KT)
      __builtin_prefetch(Hin + (size_t)(rowBase + (t >> 1)) * HDIM + kBase + 32, 0, 0);
    if (wid == 0) wait_tensor0();
    wait_async0();
    __syncthreads();
    mma_tiles(Als, Bls, lane, waveM, waveN, acc);
    __syncthreads();
  }

  // Epilogue: bias, scale by combine weight, accumulate into Out.
#pragma unroll
  for (int m = 0; m < 2; ++m)
#pragma unroll
    for (int n = 0; n < 4; ++n) {
      int col = colBase + waveN * 64 + n * 16 + (lane & 15);
      float b = bias[col];
#pragma unroll
      for (int i = 0; i < 8; ++i) {
        int row = rowBase + waveM * 32 + m * 16 + i + ((lane >> 4) << 3);
        float y = acc[m][n][i] + b;
        float cw = combine[(size_t)row * NEXP + expert];
        float val = cw * y;
        size_t o = (size_t)row * DMODEL + col;
        if (accum) val += Out[o];
        Out[o] = val;
      }
    }
}

// ---------------------------------------------------------------------------
extern "C" void kernel_launch(void* const* d_in, const int* in_sizes, int n_in,
                              void* d_out, int out_size, void* d_ws, size_t ws_size,
                              hipStream_t stream) {
  const float* inp    = (const float*)d_in[0];
  const float* gate_w = (const float*)d_in[1];
  const float* gate_b = (const float*)d_in[2];
  const float* w1     = (const float*)d_in[3];
  const float* b1     = (const float*)d_in[4];
  const float* w2     = (const float*)d_in[5];
  const float* b2     = (const float*)d_in[6];
  // d_in[7] = top_k (always 2 here)
  float* out = (float*)d_out;

  // Workspace layout (bytes):
  char* ws = (char*)d_ws;
  float*          combine = (float*)ws;                       ws += (size_t)N_TOK * NEXP * 4;   // 128 KB
  unsigned short* xbuf    = (unsigned short*)ws;              ws += (size_t)N_TOK * DMODEL * 2; // 4 MB
  unsigned short* hbuf    = (unsigned short*)ws;              ws += (size_t)N_TOK * HDIM * 2;   // 16 MB
  unsigned short* w1t     = (unsigned short*)ws;              ws += (size_t)NEXP * DMODEL * HDIM * 2; // 16 MB
  unsigned short* w2t     = (unsigned short*)ws;              // 16 MB

  // One-time precision/layout prep (amortized over 8 experts x 32 M-tiles).
  convert_x<<<(N_TOK * DMODEL) / (256 * 8), 256, 0, stream>>>(inp, xbuf);
  transpose_cvt<<<dim3(HDIM / 32, DMODEL / 32, NEXP), 256, 0, stream>>>(w1, w1t, DMODEL, HDIM);
  transpose_cvt<<<dim3(DMODEL / 32, HDIM / 32, NEXP), 256, 0, stream>>>(w2, w2t, HDIM, DMODEL);

  moe_gate<<<N_TOK / 8, 256, 0, stream>>>(inp, gate_w, gate_b, combine);

  for (int e = 0; e < NEXP; ++e) {
    moe_ffn1<<<dim3(HDIM / 128, N_TOK / 128), 256, 0, stream>>>(
        xbuf, w1t + (size_t)e * DMODEL * HDIM, b1 + (size_t)e * HDIM, hbuf);
    moe_ffn2<<<dim3(DMODEL / 128, N_TOK / 128), 256, 0, stream>>>(
        hbuf, w2t + (size_t)e * HDIM * DMODEL, b2 + (size_t)e * DMODEL,
        combine, e, (e > 0) ? 1 : 0, out);
  }
}
